// AdditiveAttention_38199439131216
// MI455X (gfx1250) — compile-verified, matrix operand order checked by reading, not run on hardware
//
#include <hip/hip_runtime.h>
#include <hip/hip_bf16.h>

// ---------------------------------------------------------------------------
// Additive attention, HIDDEN=512, batch=32, seq=4096.
// Hot path: value @ Wv via v_wmma_f32_16x16x32_bf16, tanh+Wo fused in-register.
// M=32 per wave so every B fragment feeds two WMMAs.
// ---------------------------------------------------------------------------

#define BATCH 32
#define SEQ   4096
#define HID   512

typedef __attribute__((ext_vector_type(16))) __bf16 v16bf;
typedef __attribute__((ext_vector_type(8)))  float  v8f;

union BFrag {
    v16bf v;
    uint4 q[2];
    unsigned short s[16];
};

__device__ __forceinline__ unsigned short bf16bits(float f) {
    union { float f; unsigned u; } a; a.f = f;
    unsigned u = a.u;
    u += 0x7FFFu + ((u >> 16) & 1u);      // round-to-nearest-even
    return (unsigned short)(u >> 16);
}
__device__ __forceinline__ unsigned pack2bf(float lo, float hi) {
    return (unsigned)bf16bits(lo) | ((unsigned)bf16bits(hi) << 16);
}

__device__ __forceinline__ float fast_tanh(float x) {
#if __has_builtin(__builtin_amdgcn_tanhf)
    return __builtin_amdgcn_tanhf(x);          // native v_tanh_f32
#elif __has_builtin(__builtin_amdgcn_tanh_f32)
    return __builtin_amdgcn_tanh_f32(x);
#else
    return tanhf(x);
#endif
}

// ---------------------------------------------------------------------------
// 1) Pack Wv (f32 [K=512][N=512] row-major) into bf16 WMMA B-fragments.
// Fragment layout mirrors the 16-bit A layout: lane = kGroup*16 + (n%16),
// elems 0..7 -> k = kg*8+0..7, elems 8..15 -> k = 16+kg*8+0..7 (within kstep).
// Index: WvP[((nTile*16 + kStep)*32 + lane)*16 + e]
// ---------------------------------------------------------------------------
__global__ __launch_bounds__(256)
void pack_wv_kernel(const float* __restrict__ Wv, unsigned short* __restrict__ WvP) {
    const int idx  = blockIdx.x * 256 + threadIdx.x;   // 32*16*32 = 16384 total
    const int lane = idx & 31;
    const int ks   = (idx >> 5) & 15;
    const int nt   = idx >> 9;
    const int n    = nt * 16 + (lane & 15);
    const int kg   = lane >> 4;
    unsigned short* dst = WvP + (size_t)idx * 16;
#pragma unroll
    for (int e = 0; e < 16; ++e) {
        const int k = ks * 32 + ((e < 8) ? (kg * 8 + e) : (16 + kg * 8 + (e - 8)));
        dst[e] = bf16bits(Wv[k * HID + n]);
    }
}

// ---------------------------------------------------------------------------
// 2) qadd[b][n] = (q[b] @ Wq)[n] + bq[n] + bv[n]
// ---------------------------------------------------------------------------
__global__ __launch_bounds__(256)
void qadd_kernel(const float* __restrict__ query, const float* __restrict__ Wq,
                 const float* __restrict__ bq, const float* __restrict__ bv,
                 float* __restrict__ qadd) {
    __shared__ float qs[HID];
    const int b = blockIdx.x;
    const int t = threadIdx.x;
    qs[t]       = query[b * HID + t];
    qs[t + 256] = query[b * HID + t + 256];
    __syncthreads();
#pragma unroll
    for (int j = 0; j < 2; ++j) {
        const int n = t + j * 256;
        float acc = bq[n] + bv[n];
        for (int k = 0; k < HID; ++k)
            acc = fmaf(qs[k], Wq[k * HID + n], acc);
        qadd[b * HID + n] = acc;
    }
}

// ---------------------------------------------------------------------------
// 3) Fused score kernel: one block = (batch b, 32 seq rows).
//    8 waves; wave w owns hidden[32 x 64] (cols w*64..w*64+63):
//    16 ksteps x (2 A-frags x 4 B-frags) = 8 WMMAs/kstep, B reused across M.
//    Epilogue fuses tanh(+qadd) and the @Wo reduction -> 32 scores/block.
// ---------------------------------------------------------------------------
#define LDSA_STRIDE 520   // 512 + 8 bf16 pad -> conflict-free b128 LDS reads

__global__ __launch_bounds__(256)
void score_kernel(const float* __restrict__ value,
                  const unsigned short* __restrict__ WvP,
                  const float* __restrict__ qadd,
                  const float* __restrict__ Wo,
                  float* __restrict__ scores) {
    __shared__ __align__(16) unsigned short ldsA[32 * LDSA_STRIDE];
    __shared__ float ldsScore[32];

    const int b       = blockIdx.x >> 7;          // 128 seq-blocks per batch
    const int seqBase = (blockIdx.x & 127) * 32;
    const int t    = threadIdx.x;
    const int lane = t & 31;
    const int w    = t >> 5;

    if (t < 32) ldsScore[t] = 0.0f;

    // ---- stage 32x512 f32 value tile into LDS as bf16 -------------------
    {
        const int row = t >> 3;            // 0..31
        const int kk  = (t & 7) * 64;      // 0..448
        const float* src = value + (((size_t)b * SEQ + seqBase + row) * HID + kk);
        unsigned* dst = (unsigned*)&ldsA[row * LDSA_STRIDE + kk];
#pragma unroll
        for (int i = 0; i < 16; ++i) {
            float4 f = ((const float4*)src)[i];
            dst[i * 2 + 0] = pack2bf(f.x, f.y);
            dst[i * 2 + 1] = pack2bf(f.z, f.w);
        }
    }
    __syncthreads();

    // ---- K loop ---------------------------------------------------------
    v8f acc[2][4];
    const v8f vzero = {};
#pragma unroll
    for (int mt = 0; mt < 2; ++mt)
#pragma unroll
        for (int nt = 0; nt < 4; ++nt) acc[mt][nt] = vzero;

    const int rowA = lane & 15;
    const int kg   = lane >> 4;
    const unsigned short* aBase0 = &ldsA[rowA * LDSA_STRIDE + kg * 8];
    const unsigned short* aBase1 = aBase0 + 16 * LDSA_STRIDE;
    const uint4* bBase = (const uint4*)WvP;   // [(nTile*16+ks)*32+lane]*2 uint4s

    for (int ks = 0; ks < 16; ++ks) {
        BFrag a0, a1;
        a0.q[0] = *(const uint4*)(aBase0 + ks * 32);
        a0.q[1] = *(const uint4*)(aBase0 + ks * 32 + 16);
        a1.q[0] = *(const uint4*)(aBase1 + ks * 32);
        a1.q[1] = *(const uint4*)(aBase1 + ks * 32 + 16);
#pragma unroll
        for (int nt = 0; nt < 4; ++nt) {
            BFrag bf;
            const uint4* bp = bBase + (size_t)((((w * 4 + nt) * 16 + ks) * 32 + lane)) * 2;
            bf.q[0] = bp[0];
            bf.q[1] = bp[1];
            acc[0][nt] = __builtin_amdgcn_wmma_f32_16x16x32_bf16(
                false, a0.v, false, bf.v, (short)0, acc[0][nt], false, false);
            acc[1][nt] = __builtin_amdgcn_wmma_f32_16x16x32_bf16(
                false, a1.v, false, bf.v, (short)0, acc[1][nt], false, false);
        }
    }

    // ---- epilogue: tanh(acc + qadd) dotted with Wo ----------------------
    // C layout: VGPR r, lanes 0-15 -> (M=r, N=lane); lanes 16-31 -> (M=8+r, N=lane-16)
    const int nLane = lane & 15;
    float qa[4], wo[4];
#pragma unroll
    for (int nt = 0; nt < 4; ++nt) {
        const int n = w * 64 + nt * 16 + nLane;
        qa[nt] = qadd[b * HID + n];
        wo[nt] = Wo[n];
    }

#pragma unroll
    for (int mt = 0; mt < 2; ++mt) {
        float part[8];
#pragma unroll
        for (int r = 0; r < 8; ++r) part[r] = 0.0f;
#pragma unroll
        for (int nt = 0; nt < 4; ++nt)
#pragma unroll
            for (int r = 0; r < 8; ++r) {
                const float h = fast_tanh(acc[mt][nt][r] + qa[nt]);
                part[r] = fmaf(h, wo[nt], part[r]);
            }
        // reduce over the 16 lanes sharing the same M (N dimension)
#pragma unroll
        for (int r = 0; r < 8; ++r) {
            float p = part[r];
            p += __shfl_xor(p, 1, 32);
            p += __shfl_xor(p, 2, 32);
            p += __shfl_xor(p, 4, 32);
            p += __shfl_xor(p, 8, 32);
            part[r] = p;
        }
        if ((lane & 15) == 0) {
            const int mBase = mt * 16 + (lane >> 4) * 8;
#pragma unroll
            for (int r = 0; r < 8; ++r)
                atomicAdd(&ldsScore[mBase + r], part[r]);   // ds_add_f32 across waves
        }
    }
    __syncthreads();
    if (t < 32)
        scores[(size_t)b * SEQ + seqBase + t] = ldsScore[t];
}

// ---------------------------------------------------------------------------
// 4) softmax over seq per batch (in place), adds bo first
// ---------------------------------------------------------------------------
__global__ __launch_bounds__(256)
void softmax_kernel(float* __restrict__ scores, const float* __restrict__ bo) {
    __shared__ float red[256];
    const int b = blockIdx.x;
    const int t = threadIdx.x;
    float* sc = scores + (size_t)b * SEQ;
    const float bias = bo[0];

    float v[16];
    float mx = -3.0e38f;
#pragma unroll
    for (int i = 0; i < 16; ++i) {
        v[i] = sc[t + i * 256] + bias;
        mx = fmaxf(mx, v[i]);
    }
    red[t] = mx; __syncthreads();
    for (int s2 = 128; s2 > 0; s2 >>= 1) {
        if (t < s2) red[t] = fmaxf(red[t], red[t + s2]);
        __syncthreads();
    }
    mx = red[0]; __syncthreads();

    float sum = 0.0f;
#pragma unroll
    for (int i = 0; i < 16; ++i) { v[i] = __expf(v[i] - mx); sum += v[i]; }
    red[t] = sum; __syncthreads();
    for (int s2 = 128; s2 > 0; s2 >>= 1) {
        if (t < s2) red[t] += red[t + s2];
        __syncthreads();
    }
    const float inv = 1.0f / red[0];
#pragma unroll
    for (int i = 0; i < 16; ++i) sc[t + i * 256] = v[i] * inv;
}

// ---------------------------------------------------------------------------
// 5) zero output, 6) context[b][h] = sum_s align[b][s] * value[b][s][h]
// ---------------------------------------------------------------------------
__global__ void zero_out_kernel(float* __restrict__ out) {
    out[blockIdx.x * 256 + threadIdx.x] = 0.0f;
}

__global__ __launch_bounds__(256)
void context_kernel(const float* __restrict__ value,
                    const float* __restrict__ align,
                    float* __restrict__ out) {
    const int b     = blockIdx.x >> 4;
    const int sBase = (blockIdx.x & 15) * 256;
    const int h     = threadIdx.x * 2;
    const float* vb = value + ((size_t)b * SEQ + sBase) * HID + h;
    const float* ab = align + (size_t)b * SEQ + sBase;
    float ax = 0.0f, ay = 0.0f;
    for (int s = 0; s < 256; ++s) {
        const float a  = ab[s];
        const float2 v = *(const float2*)(vb + (size_t)s * HID);
        ax = fmaf(a, v.x, ax);
        ay = fmaf(a, v.y, ay);
    }
    atomicAdd(&out[b * HID + h],     ax);
    atomicAdd(&out[b * HID + h + 1], ay);
}

// ---------------------------------------------------------------------------
extern "C" void kernel_launch(void* const* d_in, const int* in_sizes, int n_in,
                              void* d_out, int out_size, void* d_ws, size_t ws_size,
                              hipStream_t stream) {
    const float* query = (const float*)d_in[0];   // [1, 32, 512]
    const float* value = (const float*)d_in[1];   // [32, 4096, 512]
    const float* Wq    = (const float*)d_in[2];   // [512, 512]
    const float* bq    = (const float*)d_in[3];   // [512]
    const float* Wv    = (const float*)d_in[4];   // [512, 512]
    const float* bv    = (const float*)d_in[5];   // [512]
    const float* Wo    = (const float*)d_in[6];   // [512, 1]
    const float* bo    = (const float*)d_in[7];   // [1]
    float* out = (float*)d_out;                   // [32, 1, 512]

    // workspace layout
    unsigned short* WvP = (unsigned short*)d_ws;                       // 512 KB
    float* qadd   = (float*)((char*)d_ws + 512 * 1024);                //  64 KB
    float* scores = (float*)((char*)d_ws + 512 * 1024 + 64 * 1024);    // 512 KB

    pack_wv_kernel<<<64, 256, 0, stream>>>(Wv, WvP);
    qadd_kernel<<<BATCH, 256, 0, stream>>>(query, Wq, bq, bv, qadd);
    score_kernel<<<BATCH * (SEQ / 32), 256, 0, stream>>>(value, WvP, qadd, Wo, scores);
    softmax_kernel<<<BATCH, 256, 0, stream>>>(scores, bo);
    zero_out_kernel<<<(BATCH * HID) / 256, 256, 0, stream>>>(out);
    context_kernel<<<BATCH * 16, 256, 0, stream>>>(value, scores, out);
}
